// SelfAttention_8486855377013
// MI455X (gfx1250) — matrix operand order repaired and checked by reading, not compiled, not run on hardware
//
#include <hip/hip_runtime.h>
#include <hip/hip_bf16.h>

// ---------------------------------------------------------------------------
// CDNA5 (gfx1250) self-attention block, bf16 WMMA (f32 accumulate).
//   kernel 1: qkv = x @ W_attn + b_attn        -> q,k,v [B,H,S,D] bf16 (ws)
//   kernel 2: flash attention per (b,h,qblock) -> y [B,S,E] bf16 (ws)
//   kernel 3: out = y @ W_proj + b_proj        -> f32 d_out
// Pure-copy LDS staging uses GLOBAL_LOAD_ASYNC_TO_LDS_B128 (ASYNCcnt).
// ---------------------------------------------------------------------------

typedef __bf16 bf16_t;
typedef __attribute__((ext_vector_type(16))) __bf16 v16bf;
typedef __attribute__((ext_vector_type(8)))  __bf16 v8bf;
typedef __attribute__((ext_vector_type(8)))  float  v8f;

#define N_HEAD  16
#define N_EMBD  1024
#define SEQ_LEN 2048
#define BATCH   4
#define HEAD_D  64

// ---- CDNA5 async global->LDS copy (ISA ch.10 / CDNA5 ISA) ---------
// GV mode: LDS address comes from the vdst VGPR, global address from a 64-bit vaddr VGPR pair.
// Flat shared-space pointers carry the LDS offset in their low 32 bits.
__device__ __forceinline__ void async_copy_b128(void* lds_dst, const void* gsrc) {
    const unsigned int       lds = (unsigned int)(size_t)lds_dst;
    const unsigned long long ga  = (unsigned long long)(size_t)gsrc;
    asm volatile("global_load_async_to_lds_b128 %0, %1, off"
                 :: "v"(lds), "v"(ga) : "memory");
}
__device__ __forceinline__ void async_wait_all() {
    asm volatile("s_wait_asynccnt 0x0" ::: "memory");
}

// ---- WMMA fragment helpers (ISA 7.12.2 layouts, wave32) --------------------

// A-matrix 16x32 bf16 (MxK), row-major in LDS with leading dim `ld` (elems).
// lanes 0-15: row M=lane, elems 0..7 = K0..7, elems 8..15 = K16..23
// lanes 16-31: row M=lane-16, elems 0..7 = K8..15, elems 8..15 = K24..31
__device__ __forceinline__ v16bf load_frag_a(const bf16_t* base, int lane, int ld) {
    const int row = lane & 15;
    const int kh  = (lane & 16) ? 8 : 0;
    const bf16_t* p = base + row * ld + kh;
    v8bf lo = *(const v8bf*)(p);
    v8bf hi = *(const v8bf*)(p + 16);
    return __builtin_shufflevector(lo, hi, 0,1,2,3,4,5,6,7,8,9,10,11,12,13,14,15);
}

// B-matrix 32x16 bf16 (KxN), stored N-major in LDS: Blds[n*ld + k].
// lanes 0-15: col N=lane, elems 0..7 = K0..7, 8..15 = K16..23 ; lanes 16-31: col N=lane-16, K8..15, K24..31
__device__ __forceinline__ v16bf load_frag_b(const bf16_t* base, int lane, int ld) {
    const int col = lane & 15;
    const int kh  = (lane & 16) ? 8 : 0;
    const bf16_t* p = base + col * ld + kh;
    v8bf lo = *(const v8bf*)(p);
    v8bf hi = *(const v8bf*)(p + 16);
    return __builtin_shufflevector(lo, hi, 0,1,2,3,4,5,6,7,8,9,10,11,12,13,14,15);
}

__device__ __forceinline__ v8f wmma_bf16(v16bf a, v16bf b, v8f c) {
    // D = A(16x32) * B(32x16) + C, f32 accumulate
    return __builtin_amdgcn_wmma_f32_16x16x32_bf16(false, a, false, b,
                                                   (short)0, c, false, false);
}

// ---------------------------------------------------------------------------
// Kernel 1: QKV GEMM.  M=8192 (B*S), N=3072, K=1024. 128x128 tile per WG.
// 8 waves: wave (wm,wn) owns 64x32 -> acc[4][2] 16x16 tiles.
// ---------------------------------------------------------------------------
__global__ __launch_bounds__(256)
void qkv_gemm_kernel(const float* __restrict__ x, const float* __restrict__ W,
                     const float* __restrict__ bias,
                     bf16_t* __restrict__ q, bf16_t* __restrict__ k,
                     bf16_t* __restrict__ v) {
    __shared__ bf16_t Al[128 * 32];   // A tile, row-major [m][k]
    __shared__ bf16_t Bl[128 * 32];   // B tile, N-major  [n][k]

    const int tid  = threadIdx.x;
    const int lane = tid & 31;
    const int w    = tid >> 5;
    const int wm   = w >> 2;          // 0..1  (row group of 64)
    const int wn   = w & 3;           // 0..3  (col group of 32)
    const int n0   = blockIdx.x * 128;
    const int m0   = blockIdx.y * 128;

    v8f acc[4][2] = {};

    for (int kt = 0; kt < N_EMBD; kt += 32) {
        __syncthreads();
        // Stage: issue ALL global loads into registers first (MLP), then
        // convert f32->bf16 and store to LDS.
        float4 xa[4], wb[4];
        #pragma unroll
        for (int u = 0; u < 4; ++u) {     // A: 128x32 f32, 4 float4 per thread
            const int i = tid + u * 256;
            const int row = i >> 3, c4 = (i & 7) * 4;
            xa[u] = *(const float4*)(x + (size_t)(m0 + row) * N_EMBD + kt + c4);
        }
        #pragma unroll
        for (int u = 0; u < 4; ++u) {     // B: 32x128 f32, 4 float4 per thread
            const int i = tid + u * 256;
            const int r = i >> 5, c4 = (i & 31) * 4;
            wb[u] = *(const float4*)(W + (size_t)(kt + r) * 3072 + n0 + c4);
        }
        #pragma unroll
        for (int u = 0; u < 4; ++u) {
            const int i = tid + u * 256;
            const int row = i >> 3, c4 = (i & 7) * 4;
            bf16_t* d = Al + row * 32 + c4;
            d[0] = (bf16_t)xa[u].x; d[1] = (bf16_t)xa[u].y;
            d[2] = (bf16_t)xa[u].z; d[3] = (bf16_t)xa[u].w;
        }
        #pragma unroll
        for (int u = 0; u < 4; ++u) {     // transpose into Bl[n][k]
            const int i = tid + u * 256;
            const int r = i >> 5, c4 = (i & 31) * 4;
            Bl[(c4 + 0) * 32 + r] = (bf16_t)wb[u].x;
            Bl[(c4 + 1) * 32 + r] = (bf16_t)wb[u].y;
            Bl[(c4 + 2) * 32 + r] = (bf16_t)wb[u].z;
            Bl[(c4 + 3) * 32 + r] = (bf16_t)wb[u].w;
        }
        if (kt + 32 < N_EMBD) {   // hint next K-slab into cache (global_prefetch_b8)
            __builtin_prefetch(x + (size_t)(m0 + (tid >> 1)) * N_EMBD + kt + 32, 0, 1);
            __builtin_prefetch(W + (size_t)(kt + 32 + (tid & 31)) * 3072 + n0, 0, 1);
        }
        __syncthreads();

        v16bf af[4], bf[2];
        #pragma unroll
        for (int i = 0; i < 4; ++i)
            af[i] = load_frag_a(Al + (wm * 64 + i * 16) * 32, lane, 32);
        #pragma unroll
        for (int j = 0; j < 2; ++j)
            bf[j] = load_frag_b(Bl + (wn * 32 + j * 16) * 32, lane, 32);
        #pragma unroll
        for (int i = 0; i < 4; ++i)
            #pragma unroll
            for (int j = 0; j < 2; ++j)
                acc[i][j] = wmma_bf16(af[i], bf[j], acc[i][j]);
    }

    // Epilogue: bias add, scatter into q/k/v [B,H,S,D] bf16
    const int colL = lane & 15;
    const int hi8  = (lane & 16) ? 8 : 0;
    #pragma unroll
    for (int i = 0; i < 4; ++i) {
        #pragma unroll
        for (int j = 0; j < 2; ++j) {
            const int gcol  = n0 + wn * 32 + j * 16 + colL;     // 0..3071
            const float bv  = bias[gcol];
            const int which = gcol >> 10;                       // 0=q 1=k 2=v
            const int e     = gcol & 1023;
            const int h     = e >> 6, d = e & 63;
            bf16_t* dst = (which == 0) ? q : (which == 1) ? k : v;
            #pragma unroll
            for (int r = 0; r < 8; ++r) {
                const int grow = m0 + wm * 64 + i * 16 + r + hi8;   // 0..8191
                const int bb   = grow >> 11, s = grow & 2047;
                const float val = acc[i][j][r] + bv;
                dst[(((size_t)(bb * N_HEAD + h)) * SEQ_LEN + s) * HEAD_D + d] =
                    (bf16_t)val;
            }
        }
    }
}

// ---------------------------------------------------------------------------
// Kernel 2: flash attention. Grid: (S/128, B*H). One wave = 16 q rows.
// Streams 64-key K/V tiles with online softmax; causal + attention_mask.
// Q and K staging are pure bf16 copies -> async global->LDS DMA.
// ---------------------------------------------------------------------------
__global__ __launch_bounds__(256)
void flash_attn_kernel(const bf16_t* __restrict__ q, const bf16_t* __restrict__ k,
                       const bf16_t* __restrict__ v, const int* __restrict__ amask,
                       bf16_t* __restrict__ y) {
    __shared__ bf16_t Ql[128 * 64];       // Q block     [qrow][d]
    __shared__ bf16_t Kl[64 * 64];        // K tile      [key][d]   (B N-major for QK^T)
    __shared__ bf16_t Vt[64 * 64];        // V^T tile    [d][key]   (B N-major for P*V)
    __shared__ bf16_t Pl[8 * 16 * 64];    // per-wave P bounce (C-layout -> A-layout)

    const int tid  = threadIdx.x;
    const int lane = tid & 31;
    const int w    = tid >> 5;
    const int q0   = blockIdx.x * 128;
    const int bh   = blockIdx.y;              // b*H + h
    const int b    = bh >> 4;
    const int h    = bh & 15;

    const bf16_t* qbh = q + (size_t)bh * SEQ_LEN * HEAD_D;
    const bf16_t* kbh = k + (size_t)bh * SEQ_LEN * HEAD_D;
    const bf16_t* vbh = v + (size_t)bh * SEQ_LEN * HEAD_D;
    const int*    am  = amask + b * SEQ_LEN;

    // Load Q block via async DMA (128x64 bf16, 4x16B chunks per thread)
    #pragma unroll
    for (int u = 0; u < 4; ++u) {
        const int i = tid + u * 256;
        const int row = i >> 3, c = (i & 7) * 8;
        async_copy_b128(Ql + row * 64 + c, qbh + (size_t)(q0 + row) * 64 + c);
    }
    async_wait_all();
    __syncthreads();

    const v16bf qf0 = load_frag_a(Ql + w * 16 * 64,      lane, 64);  // d 0..31
    const v16bf qf1 = load_frag_a(Ql + w * 16 * 64 + 32, lane, 64);  // d 32..63

    v8f   o[4] = {};
    float mrow[8], lrow[8];
    #pragma unroll
    for (int j = 0; j < 8; ++j) { mrow[j] = -3.0e38f; lrow[j] = 0.0f; }

    const int colL = lane & 15;
    const int hi8  = (lane & 16) ? 8 : 0;
    const int kblocks = (q0 + 128) / 64;       // causal upper bound

    for (int kb = 0; kb < kblocks; ++kb) {
        __syncthreads();   // all waves done reading previous K/V tiles
        const bf16_t* kp = kbh + (size_t)kb * 64 * 64;
        const bf16_t* vp = vbh + (size_t)kb * 64 * 64;
        // K: straight copy via async DMA (64x64 bf16, 2x16B per thread)
        #pragma unroll
        for (int u = 0; u < 2; ++u) {
            const int i = tid + u * 256;
            const int row = i >> 3, c = (i & 7) * 8;
            async_copy_b128(Kl + row * 64 + c, kp + row * 64 + c);
        }
        // V: vector row reads, transpose into Vt[d][key] with scalar LDS stores
        #pragma unroll
        for (int u = 0; u < 2; ++u) {
            const int i = tid + u * 256;
            const int row = i >> 3, c = (i & 7) * 8;
            uint4 raw = *(const uint4*)(vp + (size_t)row * 64 + c);
            const bf16_t* e = (const bf16_t*)&raw;
            #pragma unroll
            for (int t = 0; t < 8; ++t)
                Vt[(c + t) * 64 + row] = e[t];
        }
        async_wait_all();
        __syncthreads();

        // S = Q K^T  (K-dim = d = 64 -> two WMMAs per 16-key tile)
        v8f s[4];
        #pragma unroll
        for (int n = 0; n < 4; ++n) {
            v8f z = {};
            v16bf b0 = load_frag_b(Kl + n * 16 * 64,      lane, 64);
            v16bf b1 = load_frag_b(Kl + n * 16 * 64 + 32, lane, 64);
            z = wmma_bf16(qf0, b0, z);
            z = wmma_bf16(qf1, b1, z);
            s[n] = z;
        }

        // mask + scale + online softmax update
        int  kcol[4]; bool kok[4];
        #pragma unroll
        for (int n = 0; n < 4; ++n) {
            kcol[n] = kb * 64 + n * 16 + colL;
            kok[n]  = (am[kcol[n]] != 0);
        }
        float mnew[8], alpha[8];
        #pragma unroll
        for (int j = 0; j < 8; ++j) {
            const int row = q0 + w * 16 + j + hi8;
            float mx = mrow[j];
            #pragma unroll
            for (int n = 0; n < 4; ++n) {
                float val = s[n][j] * 0.125f;                      // 1/sqrt(64)
                val = (kok[n] && (kcol[n] <= row)) ? val : -3.0e38f;
                s[n][j] = val;
                mx = fmaxf(mx, val);
            }
            #pragma unroll
            for (int off = 1; off < 16; off <<= 1)                 // 16-lane row max
                mx = fmaxf(mx, __shfl_xor(mx, off, 16));
            mnew[j]  = mx;
            alpha[j] = __expf(mrow[j] - mx);
            mrow[j]  = mx;
        }
        #pragma unroll
        for (int j = 0; j < 8; ++j) {
            float sum = 0.0f;
            #pragma unroll
            for (int n = 0; n < 4; ++n) {
                const float p = __expf(s[n][j] - mnew[j]);
                s[n][j] = p;
                sum += p;
            }
            #pragma unroll
            for (int off = 1; off < 16; off <<= 1)
                sum += __shfl_xor(sum, off, 16);
            lrow[j] = lrow[j] * alpha[j] + sum;
            #pragma unroll
            for (int n = 0; n < 4; ++n) o[n][j] *= alpha[j];
        }

        // C-layout P -> A-layout via wave-private LDS bounce
        bf16_t* pw = Pl + w * 16 * 64;
        #pragma unroll
        for (int n = 0; n < 4; ++n)
            #pragma unroll
            for (int j = 0; j < 8; ++j)
                pw[(j + hi8) * 64 + n * 16 + colL] = (bf16_t)s[n][j];
        // DS ops from one wave complete in order; same-array alias keeps order.
        const v16bf pf0 = load_frag_a(pw,      lane, 64);   // keys 0..31
        const v16bf pf1 = load_frag_a(pw + 32, lane, 64);   // keys 32..63

        // O += P V    (K-dim = keys = 64)
        #pragma unroll
        for (int n = 0; n < 4; ++n) {
            v16bf vb0 = load_frag_b(Vt + n * 16 * 64,      lane, 64);
            v16bf vb1 = load_frag_b(Vt + n * 16 * 64 + 32, lane, 64);
            o[n] = wmma_bf16(pf0, vb0, o[n]);
            o[n] = wmma_bf16(pf1, vb1, o[n]);
        }
    }

    // Epilogue: normalize, write y [B,S,E] bf16
    #pragma unroll
    for (int j = 0; j < 8; ++j) {
        const float inv  = (lrow[j] > 0.0f) ? (1.0f / lrow[j]) : 0.0f;
        const int   srow = q0 + w * 16 + j + hi8;
        const size_t base = ((size_t)b * SEQ_LEN + srow) * N_EMBD + h * HEAD_D + colL;
        #pragma unroll
        for (int n = 0; n < 4; ++n)
            y[base + n * 16] = (bf16_t)(o[n][j] * inv);
    }
}

// ---------------------------------------------------------------------------
// Kernel 3: output projection. M=8192, N=1024, K=1024.
// A staging is a pure bf16 copy -> async global->LDS DMA.
// ---------------------------------------------------------------------------
__global__ __launch_bounds__(256)
void proj_gemm_kernel(const bf16_t* __restrict__ y, const float* __restrict__ W,
                      const float* __restrict__ bias, float* __restrict__ out) {
    __shared__ bf16_t Al[128 * 32];
    __shared__ bf16_t Bl[128 * 32];

    const int tid  = threadIdx.x;
    const int lane = tid & 31;
    const int w    = tid >> 5;
    const int wm   = w >> 2;
    const int wn   = w & 3;
    const int n0   = blockIdx.x * 128;
    const int m0   = blockIdx.y * 128;

    v8f acc[4][2] = {};

    for (int kt = 0; kt < N_EMBD; kt += 32) {
        __syncthreads();
        // A: already bf16 -> async DMA copy (2x16B per thread)
        #pragma unroll
        for (int u = 0; u < 2; ++u) {
            const int i = tid + u * 256;
            const int row = i >> 2, c8 = (i & 3) * 8;
            async_copy_b128(Al + row * 32 + c8,
                            y + (size_t)(m0 + row) * N_EMBD + kt + c8);
        }
        // B: f32 -> bf16 with transpose; loads first, then convert+store
        float4 wb[4];
        #pragma unroll
        for (int u = 0; u < 4; ++u) {
            const int i = tid + u * 256;
            const int r = i >> 5, c4 = (i & 31) * 4;
            wb[u] = *(const float4*)(W + (size_t)(kt + r) * N_EMBD + n0 + c4);
        }
        #pragma unroll
        for (int u = 0; u < 4; ++u) {
            const int i = tid + u * 256;
            const int r = i >> 5, c4 = (i & 31) * 4;
            Bl[(c4 + 0) * 32 + r] = (bf16_t)wb[u].x;
            Bl[(c4 + 1) * 32 + r] = (bf16_t)wb[u].y;
            Bl[(c4 + 2) * 32 + r] = (bf16_t)wb[u].z;
            Bl[(c4 + 3) * 32 + r] = (bf16_t)wb[u].w;
        }
        if (kt + 32 < N_EMBD)
            __builtin_prefetch(y + (size_t)(m0 + (tid >> 1)) * N_EMBD + kt + 32, 0, 1);
        async_wait_all();
        __syncthreads();

        v16bf af[4], bf[2];
        #pragma unroll
        for (int i = 0; i < 4; ++i)
            af[i] = load_frag_a(Al + (wm * 64 + i * 16) * 32, lane, 32);
        #pragma unroll
        for (int j = 0; j < 2; ++j)
            bf[j] = load_frag_b(Bl + (wn * 32 + j * 16) * 32, lane, 32);
        #pragma unroll
        for (int i = 0; i < 4; ++i)
            #pragma unroll
            for (int j = 0; j < 2; ++j)
                acc[i][j] = wmma_bf16(af[i], bf[j], acc[i][j]);
    }

    const int colL = lane & 15;
    const int hi8  = (lane & 16) ? 8 : 0;
    #pragma unroll
    for (int i = 0; i < 4; ++i) {
        #pragma unroll
        for (int j = 0; j < 2; ++j) {
            const int gcol = n0 + wn * 32 + j * 16 + colL;
            const float bv = bias[gcol];
            #pragma unroll
            for (int r = 0; r < 8; ++r) {
                const int grow = m0 + wm * 64 + i * 16 + r + hi8;
                out[(size_t)grow * N_EMBD + gcol] = acc[i][j][r] + bv;
            }
        }
    }
}

// ---------------------------------------------------------------------------
extern "C" void kernel_launch(void* const* d_in, const int* in_sizes, int n_in,
                              void* d_out, int out_size, void* d_ws, size_t ws_size,
                              hipStream_t stream) {
    const float* x      = (const float*)d_in[0];
    const int*   amask  = (const int*)  d_in[1];
    const float* W_attn = (const float*)d_in[2];
    const float* b_attn = (const float*)d_in[3];
    const float* W_proj = (const float*)d_in[4];
    const float* b_proj = (const float*)d_in[5];
    float* out = (float*)d_out;

    // Workspace layout (bf16): q, k, v in [B,H,S,D]; y in [B,S,E]
    const size_t per = (size_t)BATCH * N_HEAD * SEQ_LEN * HEAD_D;  // 8,388,608
    bf16_t* qws = (bf16_t*)d_ws;
    bf16_t* kws = qws + per;
    bf16_t* vws = kws + per;
    bf16_t* yws = vws + per;

    qkv_gemm_kernel<<<dim3(3072 / 128, 8192 / 128), 256, 0, stream>>>(
        x, W_attn, b_attn, qws, kws, vws);
    flash_attn_kernel<<<dim3(SEQ_LEN / 128, BATCH * N_HEAD), 256, 0, stream>>>(
        qws, kws, vws, amask, yws);
    proj_gemm_kernel<<<dim3(N_EMBD / 128, 8192 / 128), 256, 0, stream>>>(
        yws, W_proj, b_proj, out);
}
